// Encoder_82274393522442
// MI455X (gfx1250) — matrix operand order repaired and checked by reading, not compiled, not run on hardware
//
#include <hip/hip_runtime.h>

// ---------------- problem constants ----------------
#define BATCH 32
#define NPTS  8192
#define ZDIM  1024

#define WAVES 4
#define MT    2                                 // M-tiles (of 16 points) per wave
#define PTS_PER_WAVE (MT * 16)                  // 32
#define PTS_PER_BLOCK (WAVES * PTS_PER_WAVE)    // 128 points / block
#define NBLK (NPTS / PTS_PER_BLOCK)             // 64 blocks per batch
#define ROWP 272                                // padded LDS act row (bf16 elems, 544B)

// dynamic-LDS partitioning (bytes)
#define ACT_BYTES  (WAVES * PTS_PER_WAVE * ROWP * 2)   // 69,632
#define PART_BYTES (WAVES * ZDIM * 4)                  // 16,384
#define WBUF_HALF_ELEMS 4096                           // 8 KB per ring slot (max KT=8 chunk)
#define WBUF_SLOTS 3                                   // TDM ring depth
#define WBUF_BYTES (WBUF_SLOTS * WBUF_HALF_ELEMS * 2)  // 24,576
#define SMEM_BYTES (ACT_BYTES + PART_BYTES + WBUF_BYTES)

// swizzled bf16 weight segment offsets (bf16 elements)
#define W1OFF 0u
#define W2OFF 2048u
#define W3OFF 10240u
#define W4OFF 43008u
#define W5OFF 108544u

typedef __attribute__((ext_vector_type(16))) __bf16       v16bf;
typedef __attribute__((ext_vector_type(8)))  float        v8f;
typedef __attribute__((ext_vector_type(4)))  unsigned int v4u;
typedef __attribute__((ext_vector_type(8)))  int          v8i_t;
typedef __attribute__((ext_vector_type(4)))  int          v4i_t;

union ABFrag { uint4 u[2]; v16bf v; };

// ---------------------------------------------------------------------------
// Tensor Data Mover: 1-D chunk (bytes <= 8192, multiple of 4) global -> LDS.
// D# packed per CDNA5 ISA:
//   group0: count=1 | lds_addr | global_addr[56:0] | type=2
//   group1: data_size=1B, tile_dim0=bytes (1 row), tensor_dim0 oversized
// Groups 2/3 unused (<=2D tensor); 6-arg builtin (clang-23 arity).
// ---------------------------------------------------------------------------
__device__ __forceinline__ void tdm_load_1d(void* ldsp, const void* gp,
                                            unsigned bytes, int lane)
{
#if __has_builtin(__builtin_amdgcn_tensor_load_to_lds)
    (void)lane;
    unsigned long long ga = (unsigned long long)(uintptr_t)gp;
    unsigned lds_off = (unsigned)(uintptr_t)ldsp;       // aperture low 32b = LDS byte addr
    v4u g0;
    g0.x = 1u;                                          // count=1, user descriptor
    g0.y = lds_off;                                     // bits 63:32  lds_addr
    g0.z = (unsigned)(ga & 0xffffffffu);                // bits 95:64  global_addr[31:0]
    g0.w = (unsigned)((ga >> 32) & 0x01ffffffu)         // bits 120:96 global_addr[56:32]
         | (2u << 30);                                  // bits 127:126 type=2 (image)
    v8i_t g1;
    unsigned td0 = 1u << 20;                            // tensor_dim0 (1B units), oversized
    g1[0] = 0;                                          // wg_mask=0, data_size=0 (1B)
    g1[1] = (int)((td0 & 0xffffu) << 16);               // tensor_dim0[15:0]
    g1[2] = (int)(((td0 >> 16) & 0xffffu) | (1u << 16)); // tensor_dim0[31:16] | tensor_dim1=1
    g1[3] = (int)(bytes << 16);                         // tile_dim0 = bytes
    g1[4] = 0;                                          // tile_dim1=0 (unused), tile_dim2=0
    g1[5] = (int)bytes;                                 // tensor_dim0_stride
    g1[6] = 0;
    g1[7] = 0;
    v4i_t z4 = {0, 0, 0, 0};
    v8i_t z8 = {0, 0, 0, 0, 0, 0, 0, 0};
    __builtin_amdgcn_tensor_load_to_lds(g0, g1, z4, z4, z8, 0);
#else
    // Fallback: wave-cooperative copy (keeps source compiling on any toolchain)
    for (unsigned off = (unsigned)lane * 16u; off < bytes; off += 32u * 16u)
        *(uint4*)((char*)ldsp + off) = *(const uint4*)((const char*)gp + off);
#endif
}

template <int N>
__device__ __forceinline__ void tdm_wait_n()
{
#if __has_builtin(__builtin_amdgcn_s_wait_tensorcnt)
    __builtin_amdgcn_s_wait_tensorcnt(N);
#endif
}

// ---------------------------------------------------------------------------
// Weight swizzle: W[Cout][Cin] f32 -> bf16 WMMA B-fragments (32x16 per frag,
// lane%16 = N col, 16 K values per lane, k = 8h+s / 16+8h+s-8). Frag f = nt*KT + kt.
// ---------------------------------------------------------------------------
__global__ void swizzle_w(const float* __restrict__ W, __bf16* __restrict__ dst,
                          int Cin, int KT, int Cout)
{
    int idx = blockIdx.x * blockDim.x + threadIdx.x;
    int total = KT * 32 * Cout;
    if (idx >= total) return;
    int f  = idx >> 9;
    int e  = idx & 511;
    int ln = e >> 4;
    int j  = e & 15;
    int nt = f / KT;
    int kt = f - nt * KT;
    int k  = kt * 32 + ((j & 8) << 1) + ((ln >> 4) << 3) + (j & 7);
    int n  = nt * 16 + (ln & 15);
    float v = (k < Cin) ? W[n * Cin + k] : 0.0f;
    dst[idx] = (__bf16)v;
}

// ---------------------------------------------------------------------------
// One fused layer, TDM-pipelined (3-slot ring, prefetch distance 2):
//  - A fragments (this wave's 32 points) register-resident from LDS act tile
//  - all KT B fragments preloaded to registers per tile (single DS wait,
//    back-to-back WMMA chain)
//  - wave0 issues one TDM per chunk; per-iteration wait is tensorcnt<=1 so
//    the newest DMA spans two compute phases (drain to 0 at the tail)
// ---------------------------------------------------------------------------
template <int KT, int NT, bool RELU, bool FINAL>
__device__ __forceinline__ void layer_pipe(__bf16* act,
                                           const __bf16* __restrict__ wseg,
                                           const float* __restrict__ bias,
                                           __bf16* wbuf, float* part,
                                           int lane, int wv)
{
    const int m    = lane & 15;
    const int hsel = lane >> 4;
    const int mo   = hsel * 8;
    const unsigned CHUNK = KT * 1024u;          // bytes per NT chunk (KT frags)

    if (wv == 0) {
        tdm_load_1d(wbuf, wseg, CHUNK, lane);
        if (NT > 1)
            tdm_load_1d(wbuf + WBUF_HALF_ELEMS, wseg + (size_t)KT * 512, CHUNK, lane);
    }

    // A fragments: lane<16 -> K 0..7 & 16..23 ; lane>=16 -> K 8..15 & 24..31
    v16bf a[MT][KT];
#pragma unroll
    for (int mt = 0; mt < MT; ++mt) {
        const __bf16* row = act + (mt * 16 + m) * ROWP + hsel * 8;
#pragma unroll
        for (int kt = 0; kt < KT; ++kt) {
            ABFrag t;
            t.u[0] = *(const uint4*)(row + kt * 32);
            t.u[1] = *(const uint4*)(row + kt * 32 + 16);
            a[mt][kt] = t.v;
        }
    }

    if (wv == 0) {                              // chunk 0 resident (oldest of <=2)
        if (NT > 1) tdm_wait_n<1>(); else tdm_wait_n<0>();
    }
    __syncthreads();

    int cur = 0;                                // ring slot for chunk nt
#pragma unroll 2
    for (int nt = 0; nt < NT; ++nt) {
        __bf16* wb = wbuf + cur * WBUF_HALF_ELEMS;
        if (wv == 0 && (nt + 2) < NT) {
            int nxt = cur + 2; if (nxt >= WBUF_SLOTS) nxt -= WBUF_SLOTS;
            tdm_load_1d(wbuf + nxt * WBUF_HALF_ELEMS,
                        wseg + (size_t)(nt + 2) * KT * 512, CHUNK, lane);
        }

        // preload all KT B fragments (DS clause, one wait), then WMMA chain
        v16bf bfr[KT];
#pragma unroll
        for (int kt = 0; kt < KT; ++kt) {
            const __bf16* bp = wb + kt * 512 + lane * 16;
            ABFrag t;
            t.u[0] = *(const uint4*)(bp);
            t.u[1] = *(const uint4*)(bp + 8);
            bfr[kt] = t.v;
        }
        v8f c[MT] = {};
#pragma unroll
        for (int kt = 0; kt < KT; ++kt)
#pragma unroll
            for (int mt = 0; mt < MT; ++mt)
                c[mt] = __builtin_amdgcn_wmma_f32_16x16x32_bf16(
                            false, a[mt][kt], false, bfr[kt], (short)0, c[mt],
                            false, false);

        float bv = bias[nt * 16 + m];
        if (FINAL) {
            float mx = -__builtin_inff();
#pragma unroll
            for (int mt = 0; mt < MT; ++mt)
#pragma unroll
                for (int r = 0; r < 8; ++r) mx = fmaxf(mx, c[mt][r] + bv);
            mx = fmaxf(mx, __shfl_xor(mx, 16, 32));    // fold M halves
            if (lane < 16) part[nt * 16 + lane] = mx;  // wave max over its 32 pts
        } else {
#pragma unroll
            for (int mt = 0; mt < MT; ++mt)
#pragma unroll
                for (int r = 0; r < 8; ++r) {
                    float h = c[mt][r] + bv;
                    if (RELU) h = fmaxf(h, 0.0f);
                    act[(mt * 16 + r + mo) * ROWP + nt * 16 + m] = (__bf16)h;
                }
        }

        // chunk nt+1 must be resident for next iteration; allow the distance-2
        // prefetch to stay in flight, drain fully at the tail
        if (wv == 0) {
            if ((nt + 2) < NT) tdm_wait_n<1>(); else tdm_wait_n<0>();
        }
        __syncthreads();

        cur = cur + 1; if (cur >= WBUF_SLOTS) cur = 0;
    }
}

// ---------------------------------------------------------------------------
// Fused encoder: 4 waves x 32 points; TDM-staged weights; all 5 layers fused.
// ---------------------------------------------------------------------------
__global__ void __launch_bounds__(128)
pointnet_fused(const float* __restrict__ x,
               const __bf16* __restrict__ wq,
               const float* __restrict__ b1, const float* __restrict__ b2,
               const float* __restrict__ b3, const float* __restrict__ b4,
               const float* __restrict__ b5,
               float* __restrict__ part_out)
{
    extern __shared__ __align__(16) char smem_raw[];
    __bf16* s_act  = (__bf16*)smem_raw;                                // [4][32][ROWP]
    float*  s_part = (float*)(smem_raw + ACT_BYTES);                   // [4][ZDIM]
    __bf16* s_wbuf = (__bf16*)(smem_raw + ACT_BYTES + PART_BYTES);     // [3][4096]

    const int tid  = threadIdx.x;
    const int wv   = tid >> 5;
    const int lane = tid & 31;
    const int b    = blockIdx.y;
    const int p0   = (blockIdx.x * WAVES + wv) * PTS_PER_WAVE;

    __bf16* act = s_act + wv * PTS_PER_WAVE * ROWP;

    // zero-pad layer-1 K slice (cols 0..31) and drop in x (3 channels, bf16)
    {
        uint4 z = make_uint4(0u, 0u, 0u, 0u);
        __bf16* row = act + lane * ROWP;       // lane == point row (32 rows)
        *(uint4*)(row)      = z;
        *(uint4*)(row + 8)  = z;
        *(uint4*)(row + 16) = z;
        *(uint4*)(row + 24) = z;
        const int p = p0 + lane;
        const float* xp = x + (size_t)b * 3 * NPTS + p;
        row[0] = (__bf16)xp[0];
        row[1] = (__bf16)xp[NPTS];
        row[2] = (__bf16)xp[2 * NPTS];
    }

    layer_pipe<1, 4,  true,  false>(act, wq + W1OFF, b1, s_wbuf, nullptr, lane, wv);
    layer_pipe<2, 8,  true,  false>(act, wq + W2OFF, b2, s_wbuf, nullptr, lane, wv);
    layer_pipe<4, 16, true,  false>(act, wq + W3OFF, b3, s_wbuf, nullptr, lane, wv);
    layer_pipe<8, 16, true,  false>(act, wq + W4OFF, b4, s_wbuf, nullptr, lane, wv);
    layer_pipe<8, 64, false, true >(act, wq + W5OFF, b5, s_wbuf,
                                    s_part + wv * ZDIM, lane, wv);

    __syncthreads();
    float* dst = part_out + ((size_t)b * NBLK + blockIdx.x) * ZDIM;
    for (int i = tid; i < ZDIM; i += blockDim.x) {
        float m0 = fmaxf(s_part[0 * ZDIM + i], s_part[1 * ZDIM + i]);
        float m1 = fmaxf(s_part[2 * ZDIM + i], s_part[3 * ZDIM + i]);
        dst[i] = fmaxf(m0, m1);
    }
}

// Deterministic final reduction over the per-block partials.
__global__ void reduce_max(const float* __restrict__ part, float* __restrict__ out)
{
    int i = blockIdx.x * blockDim.x + threadIdx.x;
    if (i >= BATCH * ZDIM) return;
    int b = i / ZDIM, z = i - b * ZDIM;
    const float* p = part + (size_t)b * NBLK * ZDIM + z;
    float m = -__builtin_inff();
#pragma unroll 8
    for (int k = 0; k < NBLK; ++k) m = fmaxf(m, p[(size_t)k * ZDIM]);
    out[i] = m;
}

// ---------------------------------------------------------------------------
extern "C" void kernel_launch(void* const* d_in, const int* in_sizes, int n_in,
                              void* d_out, int out_size, void* d_ws, size_t ws_size,
                              hipStream_t stream)
{
    const float* x  = (const float*)d_in[0];
    const float* W1 = (const float*)d_in[1];  const float* b1 = (const float*)d_in[2];
    const float* W2 = (const float*)d_in[3];  const float* b2 = (const float*)d_in[4];
    const float* W3 = (const float*)d_in[5];  const float* b3 = (const float*)d_in[6];
    const float* W4 = (const float*)d_in[7];  const float* b4 = (const float*)d_in[8];
    const float* W5 = (const float*)d_in[9];  const float* b5 = (const float*)d_in[10];
    float* out = (float*)d_out;

    char*   ws   = (char*)d_ws;
    float*  part = (float*)ws;                                   // 32*64*1024 f32 = 8 MB
    const size_t WBASE = (size_t)BATCH * NBLK * ZDIM * sizeof(float);
    __bf16* wq = (__bf16*)(ws + WBASE);                          // ~724 KB swizzled bf16

    struct { const float* W; unsigned off; int Cin, KT, Cout; } L[5] = {
        { W1, W1OFF,   3, 1,   64 },
        { W2, W2OFF,  64, 2,  128 },
        { W3, W3OFF, 128, 4,  256 },
        { W4, W4OFF, 256, 8,  256 },
        { W5, W5OFF, 256, 8, 1024 },
    };
    for (int i = 0; i < 5; ++i) {
        int total = L[i].KT * 32 * L[i].Cout;
        swizzle_w<<<(total + 255) / 256, 256, 0, stream>>>(
            L[i].W, wq + L[i].off, L[i].Cin, L[i].KT, L[i].Cout);
    }

    dim3 grid(NBLK, BATCH);
    pointnet_fused<<<grid, 128, SMEM_BYTES, stream>>>(x, wq, b1, b2, b3, b4, b5, part);

    reduce_max<<<(BATCH * ZDIM + 255) / 256, 256, 0, stream>>>(part, out);
}
